// OpticalConvolution_28999619182925
// MI455X (gfx1250) — matrix operand order repaired and checked by reading, not compiled, not run on hardware
//
#include <hip/hip_runtime.h>
#include <hip/hip_bf16.h>

// CDNA5 (gfx1250) implicit-GEMM 3x3 conv, bf16x3-split WMMA, two-phase:
//   kernel 1: repack weights into WMMA A-fragment order (bf16 hi/lo) in d_ws
//   kernel 2: per workgroup (512 thr = 16 wave32):
//     phase 1: cooperatively im2col + bf16-split the shared B tile (36 chunks)
//              into LDS in exact B-fragment layout (branch-free boundary via
//              clamped offset + select)
//     phase 2: each wave owns one 16-Cout slab; per K-chunk: 4x ds_load_b128
//              (B hi/lo) + 4x global b128 (A hi/lo, L2-resident) + 3x WMMA.

typedef __attribute__((ext_vector_type(16))) __bf16 v16bf;
typedef __attribute__((ext_vector_type(8)))  float  v8f;

#define CIN    128
#define HW     56
#define NPIX   (HW * HW)          // 3136 = 196 * 16
#define COUT   256
#define NBAT   16
#define NCHUNK 36                 // 9 taps * (128/32) channel chunks
#define WELEM  (16 * NCHUNK * 32 * 16)   // 294912 = weight element count

// ---------------------------------------------------------------------------
// Kernel 1: weight repack into A-fragment order, fp32 -> bf16 hi + lo.
// Element f = ((cb*36 + cc)*32 + lane)*16 + i
//   cc = tap*4 + chunk ; A-layout: M = lane&15, K = (i<8 ? i : i+8) + 8*(lane>>4)
// ---------------------------------------------------------------------------
__global__ __launch_bounds__(256)
void conv_wprepack(const float* __restrict__ wgt,
                   __bf16* __restrict__ whi, __bf16* __restrict__ wlo)
{
    const int f = blockIdx.x * 256 + threadIdx.x;      // 0 .. WELEM-1
    const int i    = f & 15;
    const int lane = (f >> 4) & 31;
    const int t2   = f >> 9;                           // cb*36 + cc
    const int cc   = t2 % NCHUNK;
    const int cb   = t2 / NCHUNK;

    const int half  = lane >> 4;
    const int m     = lane & 15;
    const int kloc  = ((i < 8) ? i : (i + 8)) + 8 * half;
    const int chunk = cc & 3;
    const int tap   = cc >> 2;
    const int ci    = chunk * 32 + kloc;
    const int cout  = cb * 16 + m;
    const int kh    = tap / 3, kw = tap % 3;

    const float v = wgt[(((size_t)cout * CIN + ci) * 3 + kh) * 3 + kw];
    const __bf16 h = (__bf16)v;                        // RNE
    whi[f] = h;
    wlo[f] = (__bf16)(v - (float)h);
}

// ---------------------------------------------------------------------------
// Kernel 2: main conv. grid = (196 pixel blocks, 16 batches), block = 512.
// ---------------------------------------------------------------------------
__global__ __launch_bounds__(512)
void OpticalConvolution_28999619182925_kernel(const float* __restrict__ x,
                                              const __bf16* __restrict__ whi,
                                              const __bf16* __restrict__ wlo,
                                              const float* __restrict__ bias,
                                              float* __restrict__ out)
{
    // B fragments for all 36 chunks, packed as ushort[cc][lane][16] (as u32 pairs)
    __shared__ __align__(32) unsigned int shHi[NCHUNK * 32 * 8];   // 36 KB
    __shared__ __align__(32) unsigned int shLo[NCHUNK * 32 * 8];   // 36 KB

    const int tid  = threadIdx.x;
    const int lane = tid & 31;
    const int wave = tid >> 5;                 // 0..15 == cout block
    const int sblk = blockIdx.x;               // 0..195
    const int b    = blockIdx.y;               // 0..15
    const int s0   = sblk * 16;

    const float* xb = x + (size_t)b * CIN * NPIX;

    // ---------------- Phase 1: build shared B fragments (once) -------------
    // 9216 element-pairs: p = cc*256 + lane_f*8 + ipair ; 512 threads x 18 iters
#pragma unroll 1
    for (int p = tid; p < NCHUNK * 32 * 8; p += 512) {
        const int cc     = p >> 8;
        const int rem    = p & 255;
        const int lane_f = rem >> 3;
        const int ipair  = rem & 7;
        const int i0     = ipair * 2;

        const int half  = lane_f >> 4;
        const int n     = lane_f & 15;
        const int chunk = cc & 3;
        const int tap   = cc >> 2;
        const int kh    = tap / 3, kw = tap % 3;

        const int s  = s0 + n;
        const int oh = s / HW, ow = s % HW;
        const int ih = oh + kh - 1;
        const int iw = ow + kw - 1;
        const bool inb = ((unsigned)ih < (unsigned)HW) && ((unsigned)iw < (unsigned)HW);

        const int ci0 = chunk * 32 + ((i0 < 8) ? i0 : (i0 + 8)) + 8 * half;   // K = (i<8 ? i : i+8) + 8*half
        // Branch-free: clamp offset to 0 when OOB, select 0 after load.
        size_t off = inb ? ((size_t)ci0 * NPIX + ih * HW + iw) : (size_t)0;
        float v0 = xb[off];
        float v1 = xb[off + (inb ? (size_t)NPIX : (size_t)0)];
        v0 = inb ? v0 : 0.0f;
        v1 = inb ? v1 : 0.0f;

        const __bf16 h0 = (__bf16)v0, h1 = (__bf16)v1;
        const __bf16 l0 = (__bf16)(v0 - (float)h0);
        const __bf16 l1 = (__bf16)(v1 - (float)h1);

        const unsigned int hp = (unsigned int)__builtin_bit_cast(unsigned short, h0) |
                                ((unsigned int)__builtin_bit_cast(unsigned short, h1) << 16);
        const unsigned int lp = (unsigned int)__builtin_bit_cast(unsigned short, l0) |
                                ((unsigned int)__builtin_bit_cast(unsigned short, l1) << 16);
        shHi[p] = hp;
        shLo[p] = lp;
    }

    __syncthreads();

    // ---------------- Phase 2: per-wave GEMM over 36 chunks ----------------
    const int cb   = wave;
    const int co0  = cb * 16;
    const int nIdx = lane & 15;
    const int half = lane >> 4;
    const int s    = s0 + nIdx;

    const __bf16* aHiP = whi + ((size_t)(cb * NCHUNK) * 32 + lane) * 16;
    const __bf16* aLoP = wlo + ((size_t)(cb * NCHUNK) * 32 + lane) * 16;
    const unsigned int* bHiP = &shHi[lane * 8];
    const unsigned int* bLoP = &shLo[lane * 8];

    v8f acc = {};
#pragma unroll 4
    for (int cc = 0; cc < NCHUNK; ++cc) {
        const v16bf ahi = *reinterpret_cast<const v16bf*>(aHiP + cc * 512);
        const v16bf alo = *reinterpret_cast<const v16bf*>(aLoP + cc * 512);
        const v16bf bhi = *reinterpret_cast<const v16bf*>(bHiP + cc * 256);
        const v16bf blo = *reinterpret_cast<const v16bf*>(bLoP + cc * 256);

        acc = __builtin_amdgcn_wmma_f32_16x16x32_bf16(false, ahi, false, bhi,
                                                      (short)0, acc, false, false);
        acc = __builtin_amdgcn_wmma_f32_16x16x32_bf16(false, ahi, false, blo,
                                                      (short)0, acc, false, false);
        acc = __builtin_amdgcn_wmma_f32_16x16x32_bf16(false, alo, false, bhi,
                                                      (short)0, acc, false, false);
    }

    // C/D layout: VGPR r holds M = r + 8*half, N = lane & 15.
#pragma unroll
    for (int r = 0; r < 8; ++r) {
        const int co = co0 + r + 8 * half;
        out[((size_t)b * COUT + co) * NPIX + s] = acc[r] + bias[co];
    }
}

extern "C" void kernel_launch(void* const* d_in, const int* in_sizes, int n_in,
                              void* d_out, int out_size, void* d_ws, size_t ws_size,
                              hipStream_t stream)
{
    (void)in_sizes; (void)n_in; (void)out_size; (void)ws_size;
    const float* tensor  = (const float*)d_in[0];   // [16,128,56,56]
    const float* weights = (const float*)d_in[1];   // [256,128,3,3]
    const float* bias    = (const float*)d_in[2];   // [256]
    float* out = (float*)d_out;                     // [16,256,56,56]

    __bf16* whi = (__bf16*)d_ws;                    // WELEM bf16
    __bf16* wlo = whi + WELEM;                      // WELEM bf16 (1.18 MB total)

    conv_wprepack<<<dim3(WELEM / 256), dim3(256), 0, stream>>>(weights, whi, wlo);

    dim3 grid(NPIX / 16, NBAT);                     // (196, 16)
    dim3 block(512);                                // 16 wave32 (one WGP, 72KB LDS)
    OpticalConvolution_28999619182925_kernel<<<grid, block, 0, stream>>>(
        tensor, whi, wlo, bias, out);
}
